// VectorQuantizer_79637283603208
// MI455X (gfx1250) — matrix operand order repaired and checked by reading, not compiled, not run on hardware
//
#include <hip/hip_runtime.h>
#include <hip/hip_bf16.h>

// ---------------- types for WMMA ----------------
typedef __bf16 bf16_t;
typedef bf16_t v16bf __attribute__((ext_vector_type(16)));
typedef float  v8f   __attribute__((ext_vector_type(8)));

// TDM descriptor vector types (amdgpu-toolchain clang-23 6-arg builtin)
typedef unsigned int v4u __attribute__((ext_vector_type(4)));
typedef int          v8i __attribute__((ext_vector_type(8)));
typedef int          v4i __attribute__((ext_vector_type(4)));

union Ufrag {            // 32-byte bf16 fragment, loadable as two uint4
    v16bf v;
    uint4 q[2];
};

#define VQ_N 16384       // B*H*W
#define VQ_D 256
#define VQ_K 8192
#define VQ_NT 32                  // codes per chunk (two 16x16 WMMA tiles)
#define VQ_CHUNKS (VQ_K / VQ_NT)  // 256

// LDS pointer type (addrspace(3) value = workgroup-relative LDS byte address)
typedef __attribute__((address_space(3))) unsigned short lds_us;

// ---------------- bf16 helpers (bit-level, RNE) ----------------
__device__ __forceinline__ unsigned short f2bf(float f) {
    unsigned int u = __float_as_uint(f);
    u += 0x7FFFu + ((u >> 16) & 1u);   // round-to-nearest-even
    return (unsigned short)(u >> 16);
}
__device__ __forceinline__ float bf2f(unsigned short h) {
    return __uint_as_float(((unsigned int)h) << 16);
}

// ---------------- CDNA5 Tensor Data Mover: contiguous 16KB slab -> LDS -------
// D# for a degenerate 2D tensor: 2048 x 1 tile of 8-byte elements (16 KB).
// group0: [1:0]=count=1 | [63:32]=lds_addr | [120:64]=global_addr | [127:126]=type=2
// group1: data_size=8B, tensor_dim0=2048, tensor_dim1=1, tile_dim0=2048,
//         tile_dim1=1, tensor_dim0_stride=2048; groups 2/3 zero (<=2D tensor)
__device__ __forceinline__ void tdm_load_16k(unsigned lds_addr,
                                             const unsigned short* gsrc) {
    unsigned long long ga = (unsigned long long)(uintptr_t)gsrc;
    v4u g0;
    g0.x = 1u;                                                  // count=1 (valid)
    g0.y = lds_addr;                                            // lds_addr
    g0.z = (unsigned)(ga & 0xFFFFFFFFu);                        // global_addr lo
    g0.w = (unsigned)((ga >> 32) & 0x01FFFFFFu) | (2u << 30);   // addr hi | type=2
    v8i g1;
    g1[0] = (int)(3u << 16);                  // data_size code 3 = 8 bytes
    g1[1] = (int)(2048u << 16);               // tensor_dim0[15:0] at bits 63:48
    g1[2] = (int)(1u << 16);                  // tensor_dim0 hi=0, tensor_dim1=1
    g1[3] = (int)(2048u << 16);               // tensor_dim1 hi=0, tile_dim0=2048
    g1[4] = 1;                                // tile_dim1=1, tile_dim2=0
    g1[5] = 2048;                             // tensor_dim0_stride lo
    g1[6] = 0;                                // stride hi, tensor_dim1_stride lo
    g1[7] = 0;
    v4i zz4 = {0, 0, 0, 0};
    v8i zz8 = {0, 0, 0, 0, 0, 0, 0, 0};
    // amdgpu-toolchain 6-arg form: (g0, g1, g2, g3, extra, cpol)
    __builtin_amdgcn_tensor_load_to_lds(g0, g1, zz4, zz4, zz8, 0);
}
__device__ __forceinline__ unsigned lds_off(const unsigned short* p) {
    return (unsigned)(uintptr_t)(lds_us*)p;   // addrspace(3) byte offset
}

// ---------------- K1: split z into bf16 hi/lo, zero loss cell ----------------
__global__ __launch_bounds__(256) void vq_split_z_kernel(
    const float* __restrict__ z, unsigned short* __restrict__ zhi,
    unsigned short* __restrict__ zlo, float* __restrict__ out_loss)
{
    int i = blockIdx.x * 256 + threadIdx.x;       // N*D threads
    float x = z[i];
    unsigned short h = f2bf(x);
    zhi[i] = h;
    zlo[i] = f2bf(x - bf2f(h));
    if (i == 0) *out_loss = 0.0f;
}

// ---------------- K2: split codebook + per-code squared norms ----------------
__global__ __launch_bounds__(256) void vq_codebook_kernel(
    const float* __restrict__ cb, unsigned short* __restrict__ ehi,
    unsigned short* __restrict__ elo, float* __restrict__ esq)
{
    const int k = blockIdx.x;                     // K blocks
    const int t = threadIdx.x;                    // D threads
    size_t i = (size_t)k * VQ_D + t;
    float x = cb[i];
    unsigned short h = f2bf(x);
    ehi[i] = h;
    elo[i] = f2bf(x - bf2f(h));
    float sq = x * x;
    #pragma unroll
    for (int m = 16; m >= 1; m >>= 1) sq += __shfl_xor(sq, m, 32);
    __shared__ float sred[8];
    if ((t & 31) == 0) sred[t >> 5] = sq;
    __syncthreads();
    if (t == 0) {
        float s = 0.f;
        #pragma unroll
        for (int w = 0; w < 8; ++w) s += sred[w];
        esq[k] = s;
    }
}

// ---------------- K3: main WMMA distance/argmin kernel ----------------
// Block: 128 threads = 4 waves (1 wave/SIMD); each wave owns 32 rows (2 A-tiles),
// block covers 128 rows. Loops over all 8192 codes in 32-code chunks (2 B-tiles),
// double-buffered in LDS via TDM (tensor_load_to_lds, TENSORcnt-tracked DMA):
// wave 0 issues 2 descriptors per chunk instead of 2048 per-thread copies.
// 2x2 register tiling -> each B fragment read from LDS feeds 2 row-tiles.
// score[n][k] = e_sq[k] - 2*cross[n][k]  (z_sq constant per row -> drops out of argmin)
__global__ __launch_bounds__(128) void vq_argmin_kernel(
    const unsigned short* __restrict__ zhi, const unsigned short* __restrict__ zlo,
    const unsigned short* __restrict__ ehi, const unsigned short* __restrict__ elo,
    const float* __restrict__ esq, int* __restrict__ bestidx)
{
    // [buffer][hi/lo][32 codes * 256 dims]  -> 2*2*8192*2B = 64 KB
    __shared__ unsigned short sB[2][2][VQ_NT * VQ_D];

    const int tid  = threadIdx.x;
    const int lane = tid & 31;
    const int wave = tid >> 5;
    const int half = (lane >> 4) & 1;             // lane group 0-15 / 16-31
    const int lcol = lane & 15;

    const int rowbase = blockIdx.x * 128 + wave * 32;

    // Persistent A fragments: 2 row-tiles x 8 K-chunks of 32, hi and lo
    // (16-bit A 16x32 layout: lane<16: K = 0..15 ; lane>=16: K = 16..31, matching B)
    Ufrag ahi[2][8], alo[2][8];
    #pragma unroll
    for (int t = 0; t < 2; ++t) {
        const int arow = rowbase + t * 16 + lcol;     // A-matrix: M = lane%16
        #pragma unroll
        for (int kc = 0; kc < 8; ++kc) {
            const unsigned short* ph = zhi + (size_t)arow * VQ_D + kc * 32 + half * 16;
            ahi[t][kc].q[0] = *(const uint4*)(ph);
            ahi[t][kc].q[1] = *(const uint4*)(ph + 8);
            const unsigned short* pl = zlo + (size_t)arow * VQ_D + kc * 32 + half * 16;
            alo[t][kc].q[0] = *(const uint4*)(pl);
            alo[t][kc].q[1] = *(const uint4*)(pl + 8);
        }
    }

    // TDM staging of one 32-code chunk: two contiguous 16 KB slabs (hi, lo)
    #define STAGE_CHUNK(buf, cb0)                                           \
        do {                                                                \
            tdm_load_16k(lds_off(&sB[(buf)][0][0]),                         \
                         ehi + (size_t)(cb0) * VQ_D);                       \
            tdm_load_16k(lds_off(&sB[(buf)][1][0]),                         \
                         elo + (size_t)(cb0) * VQ_D);                       \
        } while (0)

    if (wave == 0) {                 // prologue: chunk 0 -> buffer 0
        STAGE_CHUNK(0, 0);
        __builtin_amdgcn_s_wait_tensorcnt((short)0);
    }
    __syncthreads();

    float best0[8], best1[8];
    int   bidx0[8], bidx1[8];
    #pragma unroll
    for (int j = 0; j < 8; ++j) {
        best0[j] = 3.4e38f; bidx0[j] = 0;
        best1[j] = 3.4e38f; bidx1[j] = 0;
    }

    for (int it = 0; it < VQ_CHUNKS; ++it) {
        const int cur = it & 1;
        const int cb0 = it * VQ_NT;

        if (wave == 0 && it + 1 < VQ_CHUNKS)
            STAGE_CHUNK(cur ^ 1, cb0 + VQ_NT);    // DMA-prefetch next chunk

        const unsigned short* bhb = &sB[cur][0][0];
        const unsigned short* blb = &sB[cur][1][0];
        v8f acc00 = {}, acc01 = {};   // row-tile 0 x code-tile {0,1}
        v8f acc10 = {}, acc11 = {};   // row-tile 1 x code-tile {0,1}
        #pragma unroll
        for (int kc = 0; kc < 8; ++kc) {
            // B k order used here (same as A): col N = lane%16, K = half*16 + i (contiguous)
            const int o0 = lcol * VQ_D + kc * 32 + half * 16;
            const int o1 = o0 + 16 * VQ_D;
            Ufrag bh0, bl0, bh1, bl1;
            bh0.q[0] = *(const uint4*)(bhb + o0); bh0.q[1] = *(const uint4*)(bhb + o0 + 8);
            bh1.q[0] = *(const uint4*)(bhb + o1); bh1.q[1] = *(const uint4*)(bhb + o1 + 8);
            bl0.q[0] = *(const uint4*)(blb + o0); bl0.q[1] = *(const uint4*)(blb + o0 + 8);
            bl1.q[0] = *(const uint4*)(blb + o1); bl1.q[1] = *(const uint4*)(blb + o1 + 8);
            // hi*hi + hi*lo + lo*hi per tile (~fp32-accurate cross term);
            // 4 accumulators interleaved -> 4 independent WMMA dependency chains,
            // each B fragment reused by both row-tiles (halves LDS reads per WMMA)
            acc00 = __builtin_amdgcn_wmma_f32_16x16x32_bf16(false, ahi[0][kc].v, false, bh0.v,
                                                            (short)0, acc00, false, false);
            acc01 = __builtin_amdgcn_wmma_f32_16x16x32_bf16(false, ahi[0][kc].v, false, bh1.v,
                                                            (short)0, acc01, false, false);
            acc10 = __builtin_amdgcn_wmma_f32_16x16x32_bf16(false, ahi[1][kc].v, false, bh0.v,
                                                            (short)0, acc10, false, false);
            acc11 = __builtin_amdgcn_wmma_f32_16x16x32_bf16(false, ahi[1][kc].v, false, bh1.v,
                                                            (short)0, acc11, false, false);
            acc00 = __builtin_amdgcn_wmma_f32_16x16x32_bf16(false, ahi[0][kc].v, false, bl0.v,
                                                            (short)0, acc00, false, false);
            acc01 = __builtin_amdgcn_wmma_f32_16x16x32_bf16(false, ahi[0][kc].v, false, bl1.v,
                                                            (short)0, acc01, false, false);
            acc10 = __builtin_amdgcn_wmma_f32_16x16x32_bf16(false, ahi[1][kc].v, false, bl0.v,
                                                            (short)0, acc10, false, false);
            acc11 = __builtin_amdgcn_wmma_f32_16x16x32_bf16(false, ahi[1][kc].v, false, bl1.v,
                                                            (short)0, acc11, false, false);
            acc00 = __builtin_amdgcn_wmma_f32_16x16x32_bf16(false, alo[0][kc].v, false, bh0.v,
                                                            (short)0, acc00, false, false);
            acc01 = __builtin_amdgcn_wmma_f32_16x16x32_bf16(false, alo[0][kc].v, false, bh1.v,
                                                            (short)0, acc01, false, false);
            acc10 = __builtin_amdgcn_wmma_f32_16x16x32_bf16(false, alo[1][kc].v, false, bh0.v,
                                                            (short)0, acc10, false, false);
            acc11 = __builtin_amdgcn_wmma_f32_16x16x32_bf16(false, alo[1][kc].v, false, bh1.v,
                                                            (short)0, acc11, false, false);
        }

        // C layout: VGPR j -> row (half*8 + j), lane%16 -> column
        const float es0 = esq[cb0 + lcol];          // L2-resident
        const float es1 = esq[cb0 + 16 + lcol];
        const int   c0  = cb0 + lcol;
        const int   c1  = cb0 + 16 + lcol;
        #pragma unroll
        for (int j = 0; j < 8; ++j) {
            float s00 = es0 - 2.0f * acc00[j];      // lower codes tested first:
            if (s00 < best0[j]) { best0[j] = s00; bidx0[j] = c0; }   // strict <
            float s01 = es1 - 2.0f * acc01[j];      // keeps first occurrence
            if (s01 < best0[j]) { best0[j] = s01; bidx0[j] = c1; }
            float s10 = es0 - 2.0f * acc10[j];
            if (s10 < best1[j]) { best1[j] = s10; bidx1[j] = c0; }
            float s11 = es1 - 2.0f * acc11[j];
            if (s11 < best1[j]) { best1[j] = s11; bidx1[j] = c1; }
        }

        if (wave == 0)
            __builtin_amdgcn_s_wait_tensorcnt((short)0);   // DMA for next buf landed
        __syncthreads();     // everyone done computing cur; next buffer visible
    }
    #undef STAGE_CHUNK

    // reduce argmin across the 16 lanes of each half (xor masks < 16 stay in-half);
    // tie-break to the smaller index = first occurrence (argmin semantics)
    #pragma unroll
    for (int j = 0; j < 8; ++j) {
        float b0 = best0[j]; int i0 = bidx0[j];
        float b1 = best1[j]; int i1 = bidx1[j];
        #pragma unroll
        for (int m = 1; m < 16; m <<= 1) {
            float ob0 = __shfl_xor(b0, m, 32);
            int   oi0 = __shfl_xor(i0, m, 32);
            if (ob0 < b0 || (ob0 == b0 && oi0 < i0)) { b0 = ob0; i0 = oi0; }
            float ob1 = __shfl_xor(b1, m, 32);
            int   oi1 = __shfl_xor(i1, m, 32);
            if (ob1 < b1 || (ob1 == b1 && oi1 < i1)) { b1 = ob1; i1 = oi1; }
        }
        if (lcol == 0) {
            bestidx[rowbase + half * 8 + j]      = i0;   // row-tile 0
            bestidx[rowbase + 16 + half * 8 + j] = i1;   // row-tile 1
        }
    }
}

// ---------------- K4: gather z_q, indices, exact fp32 loss ----------------
__global__ __launch_bounds__(256) void vq_gather_kernel(
    const float* __restrict__ z, const float* __restrict__ cb,
    const int* __restrict__ bestidx, float* __restrict__ out_zq,
    float* __restrict__ out_idx, float* __restrict__ out_loss)
{
    const int n = blockIdx.x;                     // N blocks
    const int t = threadIdx.x;                    // D threads
    const int k = bestidx[n];
    float zv = z[(size_t)n * VQ_D + t];
    float ev = cb[(size_t)k * VQ_D + t];
    out_zq[(size_t)n * VQ_D + t] = ev;            // z_q_st == z_q numerically
    float d = zv - ev;
    float sq = d * d;
    #pragma unroll
    for (int m = 16; m >= 1; m >>= 1) sq += __shfl_xor(sq, m, 32);
    __shared__ float sred[8];
    if ((t & 31) == 0) sred[t >> 5] = sq;
    __syncthreads();
    if (t == 0) {
        float s = 0.f;
        #pragma unroll
        for (int w = 0; w < 8; ++w) s += sred[w];
        // vq_loss = (1 + beta) * mean((z - z_q)^2), beta = 0.25
        atomicAdd(out_loss, s * (1.25f / ((float)VQ_N * (float)VQ_D)));
        out_idx[n] = (float)k;
    }
}

// ---------------- launcher ----------------
extern "C" void kernel_launch(void* const* d_in, const int* in_sizes, int n_in,
                              void* d_out, int out_size, void* d_ws, size_t ws_size,
                              hipStream_t stream) {
    const float* z  = (const float*)d_in[0];      // [16,32,32,256] f32
    const float* cb = (const float*)d_in[1];      // [8192,256] f32
    float* out = (float*)d_out;

    // workspace layout
    unsigned char* w = (unsigned char*)d_ws;
    unsigned short* zhi = (unsigned short*)w; w += (size_t)VQ_N * VQ_D * 2;  // 8 MB
    unsigned short* zlo = (unsigned short*)w; w += (size_t)VQ_N * VQ_D * 2;  // 8 MB
    unsigned short* ehi = (unsigned short*)w; w += (size_t)VQ_K * VQ_D * 2;  // 4 MB
    unsigned short* elo = (unsigned short*)w; w += (size_t)VQ_K * VQ_D * 2;  // 4 MB
    float* esq = (float*)w;                   w += (size_t)VQ_K * 4;         // 32 KB
    int*   bidx = (int*)w;                                                    // 64 KB

    float* out_zq   = out;                                   // N*D
    float* out_idx  = out + (size_t)VQ_N * VQ_D;             // N (as float)
    float* out_loss = out + (size_t)VQ_N * VQ_D + VQ_N;      // 1

    vq_split_z_kernel<<<(VQ_N * VQ_D) / 256, 256, 0, stream>>>(z, zhi, zlo, out_loss);
    vq_codebook_kernel<<<VQ_K, 256, 0, stream>>>(cb, ehi, elo, esq);
    vq_argmin_kernel<<<VQ_N / 128, 128, 0, stream>>>(zhi, zlo, ehi, elo, esq, bidx);
    vq_gather_kernel<<<VQ_N, 256, 0, stream>>>(z, cb, bidx, out_zq, out_idx, out_loss);
}
